// AtomAttentionPairBias_74852690035389
// MI455X (gfx1250) — compile-verified, matrix operand order checked by reading, not run on hardware
//
#include <hip/hip_runtime.h>

// ---------------------------------------------------------------------------
// AtomAttentionPairBias for MI455X (gfx1250, wave32, WMMA + TDM).
// Memory-bound on z (1.07 GB @ 23.3 TB/s ~ 46us floor) -> flash attention
// reading z exactly once via double-buffered tensor_load_to_lds DMA;
// all GEMMs in bf16 WMMA with f32 accumulation.
// ---------------------------------------------------------------------------

#define DI __device__ __forceinline__

typedef __bf16 bf16;
typedef bf16  v16bf __attribute__((ext_vector_type(16)));
typedef float v8f   __attribute__((ext_vector_type(8)));
typedef float f4    __attribute__((ext_vector_type(4)));
typedef unsigned int u32x4 __attribute__((ext_vector_type(4)));
typedef int i32x8 __attribute__((ext_vector_type(8)));
typedef int i32x4 __attribute__((ext_vector_type(4)));

#define LN_EPS 1e-5f
#define C_DIM  128
#define CZ_DIM 16
#define NHEAD  4
#define HDIM   32

DI float fast_rcp(float x) { return __builtin_amdgcn_rcpf(x); }
DI float sigmoidf_(float x) { return fast_rcp(1.0f + __expf(-x)); }

DI v8f zero8() {
  v8f z;
#pragma unroll
  for (int i = 0; i < 8; ++i) z[i] = 0.0f;
  return z;
}

// Load a 16x32 bf16 tile in the CDNA5 WMMA 16-bit A/B register layout.
// base is row-major with leading dimension ld; tile origin (r0, k0).
DI v16bf load_tile16(const bf16* __restrict__ base, int ld, int r0, int k0,
                     int lane) {
  int m = lane & 15, half = lane >> 4;
  const bf16* p = base + (size_t)(r0 + m) * ld + k0;
  v16bf a;
#pragma unroll
  for (int v = 0; v < 8; ++v) {
    int kb = ((v & 4) ? 16 : 0) + half * 8 + (v & 3) * 2;
    a[2 * v]     = p[kb];
    a[2 * v + 1] = p[kb + 1];
  }
  return a;
}

DI v8f wmma_bf16(v16bf a, v16bf b, v8f c) {
  return __builtin_amdgcn_wmma_f32_16x16x32_bf16(false, a, false, b, (short)0,
                                                 c, false, false);
}

// reductions across a 16-lane half (matches C-tile row residency)
DI float hmax16(float v) {
#pragma unroll
  for (int s = 1; s < 16; s <<= 1) v = fmaxf(v, __shfl_xor(v, s, 16));
  return v;
}
DI float hsum16(float v) {
#pragma unroll
  for (int s = 1; s < 16; s <<= 1) v += __shfl_xor(v, s, 16);
  return v;
}

// ---------------------------------------------------------------------------
// Tensor Data Mover: 2D tile load (rows x row_elems f32) into LDS.
// D# layout per CDNA5 ISA ch.8: group0 {count, lds_addr, global_addr, type=2},
// group1 {data_size=4B, tensor dims, tile dims, dim0 stride}.
// This toolchain exposes the 6-arg builtin (extra i32x8 descriptor group).
// ---------------------------------------------------------------------------
DI void tdm_load_2d_f32(const float* gsrc, unsigned lds_byte_addr,
                        unsigned tile_elems, unsigned tile_rows,
                        unsigned tensor_dim0, unsigned tensor_dim1,
                        unsigned stride_elems) {
  unsigned long long ga = (unsigned long long)(size_t)gsrc;
  u32x4 g0;
  g0[0] = 1u;                                            // count=1 (valid)
  g0[1] = lds_byte_addr;                                 // lds_addr
  g0[2] = (unsigned)(ga & 0xFFFFFFFFu);                  // global_addr[31:0]
  g0[3] = (unsigned)((ga >> 32) & 0x01FFFFFFu) | (2u << 30);  // [56:32]|type=2
  i32x8 g1;
  g1[0] = 0x00020000;                                    // data_size=4B
  g1[1] = (int)((tensor_dim0 & 0xFFFFu) << 16);          // tdim0[15:0]
  g1[2] = (int)(((tensor_dim0 >> 16) & 0xFFFFu) |
                ((tensor_dim1 & 0xFFFFu) << 16));        // tdim0[31:16]|tdim1
  g1[3] = (int)(((tensor_dim1 >> 16) & 0xFFFFu) |
                (tile_elems << 16));                     // tdim1[31:16]|tile0
  g1[4] = (int)(tile_rows & 0xFFFFu);                    // tile1 (tile2=0)
  g1[5] = (int)stride_elems;                             // dim0 stride[31:0]
  g1[6] = 0;                                             // stride[47:32]
  g1[7] = 0;
  i32x4 gz4;
  gz4[0] = 0; gz4[1] = 0; gz4[2] = 0; gz4[3] = 0;
  i32x8 gz8;
#pragma unroll
  for (int i = 0; i < 8; ++i) gz8[i] = 0;
  __builtin_amdgcn_tensor_load_to_lds(g0, g1, gz4, gz4, gz8, 0);
}

DI unsigned lds_addr_of(const void* p) {
  // generic LDS pointer: low 32 bits are the wave-relative LDS byte address
  return (unsigned)(size_t)p;
}

// ---------------------------------------------------------------------------
// small elementwise kernels
// ---------------------------------------------------------------------------
__global__ void wconv_kernel(const float* __restrict__ W,
                             bf16* __restrict__ WT) {
  int i = blockIdx.x * blockDim.x + threadIdx.x;
  if (i < C_DIM * C_DIM) {
    int k = i >> 7, n = i & 127;
    WT[n * C_DIM + k] = (bf16)W[k * C_DIM + n];  // store W^T in bf16
  }
}

__global__ void cvt_kernel(const float* __restrict__ X, bf16* __restrict__ Y,
                           int n) {
  int i = blockIdx.x * blockDim.x + threadIdx.x;
  if (i < n) Y[i] = (bf16)X[i];
}

// ---------------------------------------------------------------------------
// row LayerNorms for a (no affine, keep f32) and s (scale-only, emit bf16)
// ---------------------------------------------------------------------------
DI float blockReduce128(float v, float* sh) {
  int t = threadIdx.x;
  sh[t] = v;
  __syncthreads();
#pragma unroll
  for (int s = 64; s > 0; s >>= 1) {
    if (t < s) sh[t] += sh[t + s];
    __syncthreads();
  }
  float r = sh[0];
  __syncthreads();
  return r;
}

__global__ __launch_bounds__(128) void ln_prep_kernel(
    const float* __restrict__ a, const float* __restrict__ s,
    const float* __restrict__ sscale, float* __restrict__ aln,
    bf16* __restrict__ slnb) {
  __shared__ float sh[128];
  int row = blockIdx.x, t = threadIdx.x;
  float av = a[(size_t)row * C_DIM + t];
  float sv = s[(size_t)row * C_DIM + t];
  float sumA  = blockReduce128(av, sh);
  float sumA2 = blockReduce128(av * av, sh);
  float sumS  = blockReduce128(sv, sh);
  float sumS2 = blockReduce128(sv * sv, sh);
  float mA = sumA * (1.f / 128.f), vA = sumA2 * (1.f / 128.f) - mA * mA;
  float mS = sumS * (1.f / 128.f), vS = sumS2 * (1.f / 128.f) - mS * mS;
  aln[(size_t)row * C_DIM + t] = (av - mA) * rsqrtf(vA + LN_EPS);
  slnb[(size_t)row * C_DIM + t] =
      (bf16)(((sv - mS) * rsqrtf(vS + LN_EPS)) * sscale[t]);
}

// ---------------------------------------------------------------------------
// AdaLN: a' = sigmoid(sln@Wg + bg) * aln + sln@Wb     (one wave / 16 rows)
// ---------------------------------------------------------------------------
__global__ __launch_bounds__(32) void adaln_kernel(
    const bf16* __restrict__ sln, const float* __restrict__ aln,
    const bf16* __restrict__ WgT, const float* __restrict__ bg,
    const bf16* __restrict__ WbT, bf16* __restrict__ aout) {
  int lane = threadIdx.x & 31;
  int r0 = blockIdx.x * 16;
  v16bf A[4];
#pragma unroll
  for (int kc = 0; kc < 4; ++kc)
    A[kc] = load_tile16(sln, C_DIM, r0, kc * 32, lane);
#pragma unroll
  for (int nt = 0; nt < 8; ++nt) {
    v8f g = zero8(), bv = zero8();
#pragma unroll
    for (int kc = 0; kc < 4; ++kc) {
      g  = wmma_bf16(A[kc], load_tile16(WgT, C_DIM, nt * 16, kc * 32, lane), g);
      bv = wmma_bf16(A[kc], load_tile16(WbT, C_DIM, nt * 16, kc * 32, lane), bv);
    }
#pragma unroll
    for (int r = 0; r < 8; ++r) {
      int m = (lane >> 4) * 8 + r, n = nt * 16 + (lane & 15);
      float o = sigmoidf_(g[r] + bg[n]) * aln[(size_t)(r0 + m) * C_DIM + n] +
                bv[r];
      aout[(size_t)(r0 + m) * C_DIM + n] = (bf16)o;
    }
  }
}

// ---------------------------------------------------------------------------
// dual projection. mode 0 (q side): Y1 = (X@W1+b1)*scale bf16, Y2 = sigmoid
// f32 (gate). mode 1 (k side): Y1 = X@W1 bf16 row-major (K), Y2 = X@W2 bf16
// stored transposed [n][Nrows] (V^T for the PV WMMA B-operand).
// ---------------------------------------------------------------------------
__global__ __launch_bounds__(32) void proj_kernel(
    const bf16* __restrict__ X, const bf16* __restrict__ W1T,
    const float* __restrict__ b1, const bf16* __restrict__ W2T,
    bf16* __restrict__ Y1, void* __restrict__ Y2, float scale, int mode,
    int Nrows) {
  int lane = threadIdx.x & 31;
  int r0 = blockIdx.x * 16;
  v16bf A[4];
#pragma unroll
  for (int kc = 0; kc < 4; ++kc)
    A[kc] = load_tile16(X, C_DIM, r0, kc * 32, lane);
#pragma unroll
  for (int nt = 0; nt < 8; ++nt) {
    v8f a1 = zero8(), a2 = zero8();
#pragma unroll
    for (int kc = 0; kc < 4; ++kc) {
      a1 = wmma_bf16(A[kc], load_tile16(W1T, C_DIM, nt * 16, kc * 32, lane), a1);
      a2 = wmma_bf16(A[kc], load_tile16(W2T, C_DIM, nt * 16, kc * 32, lane), a2);
    }
#pragma unroll
    for (int r = 0; r < 8; ++r) {
      int m = (lane >> 4) * 8 + r, n = nt * 16 + (lane & 15);
      size_t idx = (size_t)(r0 + m) * C_DIM + n;
      if (mode == 0) {
        Y1[idx] = (bf16)((a1[r] + b1[n]) * scale);          // q, pre-scaled
        ((float*)Y2)[idx] = sigmoidf_(a2[r]);               // gate
      } else {
        Y1[idx] = (bf16)a1[r];                              // k row-major
        ((bf16*)Y2)[(size_t)n * Nrows + (r0 + m)] = (bf16)a2[r];  // v^T
      }
    }
  }
}

// ---------------------------------------------------------------------------
// flash attention, one wave per 16 q-rows, all 4 heads together so the z
// tile (pair bias) is DMA'd from HBM exactly once (TDM, double-buffered).
// ---------------------------------------------------------------------------
#define ZTILE (16 * 32 * CZ_DIM)  // 8192 f32 = 32KB per buffer

__global__ __launch_bounds__(32) void attn_kernel(
    const bf16* __restrict__ qm, const bf16* __restrict__ km,
    const bf16* __restrict__ vT, const float* __restrict__ z,
    const float* __restrict__ lnz_scale, const float* __restrict__ wbias,
    const float* __restrict__ gate, bf16* __restrict__ og, int Nq, int Nk) {
  __shared__ float zsh[2][ZTILE];
  __shared__ float sLnz[CZ_DIM];
  __shared__ float sWb[CZ_DIM * NHEAD];
  __shared__ float biasbuf[16 * 32 * NHEAD];
  __shared__ bf16 pbuf[16 * 32];

  int lane = threadIdx.x & 31;
  int q0 = blockIdx.x * 16;

  if (lane < CZ_DIM) sLnz[lane] = lnz_scale[lane];
  sWb[lane] = wbias[lane];
  sWb[lane + 32] = wbias[lane + 32];
  __syncthreads();

  const unsigned tdim0 = (unsigned)Nk * CZ_DIM;  // z row length (elems)
  // kick off first z tile DMA: 16 rows x 512 f32, row stride Nk*16
  tdm_load_2d_f32(z + (size_t)q0 * Nk * CZ_DIM, lds_addr_of(&zsh[0][0]),
                  32 * CZ_DIM, 16, tdim0, (unsigned)Nq, tdim0);

  v16bf qa[NHEAD];
#pragma unroll
  for (int h = 0; h < NHEAD; ++h)
    qa[h] = load_tile16(qm, C_DIM, q0, h * HDIM, lane);

  v8f O[NHEAD][2];
  float rm[NHEAD][8], rl[NHEAD][8];
#pragma unroll
  for (int h = 0; h < NHEAD; ++h)
#pragma unroll
    for (int r = 0; r < 8; ++r) {
      O[h][0][r] = 0.f;
      O[h][1][r] = 0.f;
      rm[h][r] = -1e30f;
      rl[h][r] = 0.f;
    }

  for (int k0 = 0; k0 < Nk; k0 += 32) {
    int cur = (k0 >> 5) & 1;
    if (k0 + 32 < Nk) {
      // overlap: DMA next z tile into the other buffer while we compute
      tdm_load_2d_f32(z + ((size_t)q0 * Nk + (k0 + 32)) * CZ_DIM,
                      lds_addr_of(&zsh[cur ^ 1][0]), 32 * CZ_DIM, 16, tdim0,
                      (unsigned)Nq, tdim0);
      __builtin_amdgcn_s_wait_tensorcnt(1);  // current tile landed (in-order)
      __builtin_prefetch(km + (size_t)(k0 + 32) * C_DIM, 0, 1);
    } else {
      __builtin_amdgcn_s_wait_tensorcnt(0);
    }

    // ---- LN(z)*lnz_scale @ Wbias for all 4 heads, 16x32 tile -> LDS ----
#pragma unroll 4
    for (int i = 0; i < 16; ++i) {
      int p = i * 32 + lane;
      int mm = p >> 5, nn = p & 31;
      const float* zp = &zsh[cur][(mm * 32 + nn) * CZ_DIM];
      f4 z0 = *(const f4*)(zp + 0);
      f4 z1 = *(const f4*)(zp + 4);
      f4 z2 = *(const f4*)(zp + 8);
      f4 z3 = *(const f4*)(zp + 12);
      float zv[16];
#pragma unroll
      for (int j = 0; j < 4; ++j) {
        zv[j] = z0[j]; zv[4 + j] = z1[j]; zv[8 + j] = z2[j]; zv[12 + j] = z3[j];
      }
      float su = 0.f, sq = 0.f;
#pragma unroll
      for (int j = 0; j < 16; ++j) { su += zv[j]; sq += zv[j] * zv[j]; }
      float mean = su * (1.f / 16.f);
      float var = sq * (1.f / 16.f) - mean * mean;
      float inv = rsqrtf(var + LN_EPS);
      float b0 = 0.f, b1 = 0.f, b2 = 0.f, b3 = 0.f;
#pragma unroll
      for (int cz = 0; cz < 16; ++cz) {
        float t = (zv[cz] - mean) * inv * sLnz[cz];
        b0 += t * sWb[cz * 4 + 0];
        b1 += t * sWb[cz * 4 + 1];
        b2 += t * sWb[cz * 4 + 2];
        b3 += t * sWb[cz * 4 + 3];
      }
      float* bb = &biasbuf[(mm * 32 + nn) * 4];
      bb[0] = b0; bb[1] = b1; bb[2] = b2; bb[3] = b3;
    }
    __syncthreads();

#pragma unroll
    for (int h = 0; h < NHEAD; ++h) {
      // scores: S = q_h @ k_h^T + bias (bias rides in as the WMMA C operand)
      v8f S[2];
#pragma unroll
      for (int t = 0; t < 2; ++t) {
        v8f c;
#pragma unroll
        for (int r = 0; r < 8; ++r) {
          int m = (lane >> 4) * 8 + r, n = t * 16 + (lane & 15);
          c[r] = biasbuf[(m * 32 + n) * 4 + h];
        }
        v16bf B = load_tile16(km, C_DIM, k0 + t * 16, h * HDIM, lane);
        S[t] = wmma_bf16(qa[h], B, c);
      }
      // online softmax over the 32 new columns
#pragma unroll
      for (int r = 0; r < 8; ++r) {
        float mx = hmax16(fmaxf(S[0][r], S[1][r]));
        float mo = rm[h][r];
        float mn = fmaxf(mo, mx);
        float alpha = __expf(mo - mn);
        float p0 = __expf(S[0][r] - mn);
        float p1 = __expf(S[1][r] - mn);
        float ls = hsum16(p0 + p1);
        rl[h][r] = rl[h][r] * alpha + ls;
        rm[h][r] = mn;
        O[h][0][r] *= alpha;
        O[h][1][r] *= alpha;
        S[0][r] = p0;
        S[1][r] = p1;
      }
      // C-layout P -> LDS -> A-layout bf16 for the PV WMMA
      __syncthreads();
#pragma unroll
      for (int t = 0; t < 2; ++t)
#pragma unroll
        for (int r = 0; r < 8; ++r) {
          int m = (lane >> 4) * 8 + r, n = t * 16 + (lane & 15);
          pbuf[m * 32 + n] = (bf16)S[t][r];
        }
      __syncthreads();
      v16bf P = load_tile16(pbuf, 32, 0, 0, lane);
#pragma unroll
      for (int t = 0; t < 2; ++t) {
        v16bf Vb = load_tile16(vT, Nk, h * HDIM + t * 16, k0, lane);
        O[h][t] = wmma_bf16(P, Vb, O[h][t]);
      }
    }
    __syncthreads();
  }

  // epilogue: normalize, gate, emit bf16 for the output projection
#pragma unroll
  for (int h = 0; h < NHEAD; ++h)
#pragma unroll
    for (int t = 0; t < 2; ++t)
#pragma unroll
      for (int r = 0; r < 8; ++r) {
        int m = (lane >> 4) * 8 + r;
        int n = h * HDIM + t * 16 + (lane & 15);
        float val = O[h][t][r] * fast_rcp(rl[h][r]);
        val *= gate[(size_t)(q0 + m) * C_DIM + n];
        og[(size_t)(q0 + m) * C_DIM + n] = (bf16)val;
      }
}

// ---------------------------------------------------------------------------
// out = sigmoid(s_q@Ws + bs) * (og@Wo)
// ---------------------------------------------------------------------------
__global__ __launch_bounds__(32) void final_kernel(
    const bf16* __restrict__ sqb, const bf16* __restrict__ WsT,
    const float* __restrict__ bs, const bf16* __restrict__ ogb,
    const bf16* __restrict__ WoT, float* __restrict__ out) {
  int lane = threadIdx.x & 31;
  int r0 = blockIdx.x * 16;
  v16bf As[4], Ao[4];
#pragma unroll
  for (int kc = 0; kc < 4; ++kc) {
    As[kc] = load_tile16(sqb, C_DIM, r0, kc * 32, lane);
    Ao[kc] = load_tile16(ogb, C_DIM, r0, kc * 32, lane);
  }
#pragma unroll
  for (int nt = 0; nt < 8; ++nt) {
    v8f a1 = zero8(), a2 = zero8();
#pragma unroll
    for (int kc = 0; kc < 4; ++kc) {
      a1 = wmma_bf16(As[kc], load_tile16(WsT, C_DIM, nt * 16, kc * 32, lane), a1);
      a2 = wmma_bf16(Ao[kc], load_tile16(WoT, C_DIM, nt * 16, kc * 32, lane), a2);
    }
#pragma unroll
    for (int r = 0; r < 8; ++r) {
      int m = (lane >> 4) * 8 + r, n = nt * 16 + (lane & 15);
      out[(size_t)(r0 + m) * C_DIM + n] = sigmoidf_(a1[r] + bs[n]) * a2[r];
    }
  }
}

// ---------------------------------------------------------------------------
extern "C" void kernel_launch(void* const* d_in, const int* in_sizes, int n_in,
                              void* d_out, int out_size, void* d_ws,
                              size_t ws_size, hipStream_t stream) {
  const float* a_q      = (const float*)d_in[0];
  const float* a_k      = (const float*)d_in[1];
  const float* zf       = (const float*)d_in[2];
  const float* s_q      = (const float*)d_in[3];
  const float* s_k      = (const float*)d_in[4];
  const float* Wg_q     = (const float*)d_in[5];
  const float* bg_q     = (const float*)d_in[6];
  const float* Wb_q     = (const float*)d_in[7];
  const float* sscale_q = (const float*)d_in[8];
  const float* Wg_k     = (const float*)d_in[9];
  const float* bg_k     = (const float*)d_in[10];
  const float* Wb_k     = (const float*)d_in[11];
  const float* sscale_k = (const float*)d_in[12];
  const float* lnz      = (const float*)d_in[13];
  const float* Wq       = (const float*)d_in[14];
  const float* bq       = (const float*)d_in[15];
  const float* Wk       = (const float*)d_in[16];
  const float* Wv       = (const float*)d_in[17];
  const float* Wbias    = (const float*)d_in[18];
  const float* Wgate    = (const float*)d_in[19];
  const float* Wo       = (const float*)d_in[20];
  const float* Ws       = (const float*)d_in[21];
  const float* bs       = (const float*)d_in[22];

  const int Nq = in_sizes[0] / C_DIM;
  const int Nk = in_sizes[1] / C_DIM;

  char* wp = (char*)d_ws;
  auto alloc = [&](size_t b) -> void* {
    void* p = (void*)wp;
    wp += (b + 255) & ~(size_t)255;
    return p;
  };
  const size_t WB = (size_t)C_DIM * C_DIM * sizeof(bf16);
  bf16* WgqT   = (bf16*)alloc(WB);
  bf16* WbqT   = (bf16*)alloc(WB);
  bf16* WgkT   = (bf16*)alloc(WB);
  bf16* WbkT   = (bf16*)alloc(WB);
  bf16* WqT    = (bf16*)alloc(WB);
  bf16* WkT    = (bf16*)alloc(WB);
  bf16* WvT    = (bf16*)alloc(WB);
  bf16* WgateT = (bf16*)alloc(WB);
  bf16* WoT    = (bf16*)alloc(WB);
  bf16* WsT    = (bf16*)alloc(WB);

  bf16*  sqb   = (bf16*)alloc((size_t)Nq * C_DIM * sizeof(bf16));
  float* alnq  = (float*)alloc((size_t)Nq * C_DIM * sizeof(float));
  float* alnk  = (float*)alloc((size_t)Nk * C_DIM * sizeof(float));
  bf16*  slnq  = (bf16*)alloc((size_t)Nq * C_DIM * sizeof(bf16));
  bf16*  slnk  = (bf16*)alloc((size_t)Nk * C_DIM * sizeof(bf16));
  bf16*  aqb   = (bf16*)alloc((size_t)Nq * C_DIM * sizeof(bf16));
  bf16*  akb   = (bf16*)alloc((size_t)Nk * C_DIM * sizeof(bf16));
  bf16*  qb    = (bf16*)alloc((size_t)Nq * C_DIM * sizeof(bf16));
  bf16*  kbm   = (bf16*)alloc((size_t)Nk * C_DIM * sizeof(bf16));
  bf16*  vTb   = (bf16*)alloc((size_t)Nk * C_DIM * sizeof(bf16));
  float* gatef = (float*)alloc((size_t)Nq * C_DIM * sizeof(float));
  bf16*  ogb   = (bf16*)alloc((size_t)Nq * C_DIM * sizeof(bf16));

  // weight convert + transpose to bf16 [n][k]
  wconv_kernel<<<64, 256, 0, stream>>>(Wg_q, WgqT);
  wconv_kernel<<<64, 256, 0, stream>>>(Wb_q, WbqT);
  wconv_kernel<<<64, 256, 0, stream>>>(Wg_k, WgkT);
  wconv_kernel<<<64, 256, 0, stream>>>(Wb_k, WbkT);
  wconv_kernel<<<64, 256, 0, stream>>>(Wq, WqT);
  wconv_kernel<<<64, 256, 0, stream>>>(Wk, WkT);
  wconv_kernel<<<64, 256, 0, stream>>>(Wv, WvT);
  wconv_kernel<<<64, 256, 0, stream>>>(Wgate, WgateT);
  wconv_kernel<<<64, 256, 0, stream>>>(Wo, WoT);
  wconv_kernel<<<64, 256, 0, stream>>>(Ws, WsT);

  cvt_kernel<<<(Nq * C_DIM + 255) / 256, 256, 0, stream>>>(s_q, sqb,
                                                           Nq * C_DIM);

  ln_prep_kernel<<<Nq, 128, 0, stream>>>(a_q, s_q, sscale_q, alnq, slnq);
  ln_prep_kernel<<<Nk, 128, 0, stream>>>(a_k, s_k, sscale_k, alnk, slnk);

  adaln_kernel<<<Nq / 16, 32, 0, stream>>>(slnq, alnq, WgqT, bg_q, WbqT, aqb);
  adaln_kernel<<<Nk / 16, 32, 0, stream>>>(slnk, alnk, WgkT, bg_k, WbkT, akb);

  const float scale = 0.17677669529663687f;  // 1/sqrt(32)
  proj_kernel<<<Nq / 16, 32, 0, stream>>>(aqb, WqT, bq, WgateT, qb,
                                          (void*)gatef, scale, 0, Nq);
  proj_kernel<<<Nk / 16, 32, 0, stream>>>(akb, WkT, nullptr, WvT, kbm,
                                          (void*)vTb, 1.0f, 1, Nk);

  attn_kernel<<<Nq / 16, 32, 0, stream>>>(qb, kbm, vTb, zf, lnz, Wbias, gatef,
                                          ogb, Nq, Nk);

  final_kernel<<<Nq / 16, 32, 0, stream>>>(sqb, WsT, bs, ogb, WoT,
                                           (float*)d_out);
}
